// LogicLayer_49855980372094
// MI455X (gfx1250) — matrix operand order repaired and checked by reading, not compiled, not run on hardware
//
#include <hip/hip_runtime.h>
#include <hip/hip_bf16.h>

#define BATCH   4096
#define IN_DIM  2048
#define OUT_DIM 8192
#define ROWS_PER_BLOCK 64

typedef __attribute__((ext_vector_type(16))) _Float16      v16h;
typedef __attribute__((ext_vector_type(8)))  float         v8f;
typedef __attribute__((ext_vector_type(4)))  float         f32x4;
typedef __attribute__((ext_vector_type(4)))  unsigned char u8x4;

// GATE_COEFFS from the reference (exact small integers; exactly representable in f16)
__constant__ float kGateCoeffs[16][4] = {
  {0,0,0,0},{0,0,0,1},{0,1,0,-1},{0,1,0,0},
  {0,0,1,-1},{0,0,1,0},{0,1,1,-2},{0,1,1,-1},
  {1,-1,-1,1},{1,-1,-1,2},{1,0,-1,0},{1,0,-1,1},
  {1,-1,0,0},{1,-1,0,1},{1,0,0,-1},{1,0,0,0}
};

// ---------------------------------------------------------------------------
// Kernel 1: masked argmax over IN_DIM for link_weights_a and link_weights_b.
// One wave32 per row; rows [0,OUT_DIM) -> matrix A, [OUT_DIM,2*OUT_DIM) -> B.
// Streams ~170 MB once with nontemporal (TH=NT) loads. First-max semantics
// match jnp.argmax (strict > within a lane's ascending scan, tie -> lower idx
// in the cross-lane reduction).
// ---------------------------------------------------------------------------
__global__ __launch_bounds__(256) void link_argmax_kernel(
    const float* __restrict__ wa, const float* __restrict__ wb,
    const unsigned char* __restrict__ ma, const unsigned char* __restrict__ mb,
    int* __restrict__ idx_a, int* __restrict__ idx_b) {
  const int lane  = threadIdx.x & 31;
  const int gwave = (blockIdx.x * blockDim.x + threadIdx.x) >> 5;
  const bool isB  = gwave >= OUT_DIM;
  const int  row  = isB ? (gwave - OUT_DIM) : gwave;

  const float*         wrow = (isB ? wb : wa) + (size_t)row * IN_DIM;
  const unsigned char* mrow = (isB ? mb : ma) + (size_t)row * IN_DIM;

  float best = -__builtin_inff();
  int   bidx = 0;
  #pragma unroll 4
  for (int it = 0; it < IN_DIM / 128; ++it) {
    const int k0 = it * 128 + lane * 4;
    const f32x4 v = __builtin_nontemporal_load((const f32x4*)(wrow + k0));
    const u8x4  m = __builtin_nontemporal_load((const u8x4*)(mrow + k0));
    const float e0 = m.x ? v.x : -__builtin_inff();
    const float e1 = m.y ? v.y : -__builtin_inff();
    const float e2 = m.z ? v.z : -__builtin_inff();
    const float e3 = m.w ? v.w : -__builtin_inff();
    if (e0 > best) { best = e0; bidx = k0;     }
    if (e1 > best) { best = e1; bidx = k0 + 1; }
    if (e2 > best) { best = e2; bidx = k0 + 2; }
    if (e3 > best) { best = e3; bidx = k0 + 3; }
  }
  // wave32 reduction, tie -> lower index (matches argmax "first" semantics)
  #pragma unroll
  for (int off = 16; off >= 1; off >>= 1) {
    const float ov = __shfl_xor(best, off, 32);
    const int   oi = __shfl_xor(bidx, off, 32);
    if (ov > best || (ov == best && oi < bidx)) { best = ov; bidx = oi; }
  }
  if (lane == 0) (isB ? idx_b : idx_a)[row] = bidx;
}

// ---------------------------------------------------------------------------
// Kernel 2: gate coefficients c = one_hot(argmax(masked nw)) @ GATE_COEFFS
// via V_WMMA_F32_16X16X32_F16. One wave handles 16 neurons.
//   A (16x32 f16): one-hot rows (K=0..15 = gates, K=16..31 = 0). Contiguous-K layout:
//     lanes 0-15 hold M=lane&15, halves 0..15 = K0..15; lanes 16-31 = K16..31 (zero).
//   B (32x16 f16): lane n<4 holds column n = GATE_COEFFS[k][n], k=0..15;
//     lanes 16-31 (K=16..31) all zero.
//   D (16x16 f32): VGPR r, lanes 0-15 -> (M=r, N=lane); lanes 16-31 -> (M=8+r).
// Exact: one-hot × small-int f16 with f32 accumulate.
// ---------------------------------------------------------------------------
__global__ __launch_bounds__(256) void gate_wmma_kernel(
    const float* __restrict__ nw, const unsigned char* __restrict__ gmask,
    float* __restrict__ coeffs /* OUT_DIM x 4 */) {
  const int lane  = threadIdx.x & 31;
  const int gwave = (blockIdx.x * blockDim.x + threadIdx.x) >> 5;
  const int nbase = gwave * 16;
  const int row   = nbase + (lane & 15);

  // masked argmax over the 16 gates (first-max semantics)
  int   gbest = 0;
  float vbest = -__builtin_inff();
  #pragma unroll
  for (int g = 0; g < 16; ++g) {
    const float w = gmask[row * 16 + g] ? nw[row * 16 + g] : -__builtin_inff();
    if (w > vbest) { vbest = w; gbest = g; }
  }

  // Build A: one-hot in the wave-striped 16x32 f16 layout
  const int kofs = (lane < 16) ? 0 : 16;
  v16h A = {};
  #pragma unroll
  for (int h = 0; h < 16; ++h)
    A[h] = (_Float16)((gbest == kofs + h) ? 1.0f : 0.0f);

  // Build B: GATE_COEFFS^ (32x16), only k<16 and n<4 nonzero
  v16h B = {};
  if (lane < 4) {
    #pragma unroll
    for (int k = 0; k < 16; ++k)
      B[k] = (_Float16)kGateCoeffs[k][lane];
  }

  v8f C = {};
  C = __builtin_amdgcn_wmma_f32_16x16x32_f16(
      /*neg_a=*/false, A, /*neg_b=*/false, B,
      /*c_mod=*/(short)0, C, /*reuse_a=*/false, /*reuse_b=*/false);

  // Scatter D columns 0..3 back as float4-per-neuron coefficient rows
  const int n = lane & 15;
  if (n < 4) {
    const int mofs = (lane < 16) ? 0 : 8;
    #pragma unroll
    for (int r = 0; r < 8; ++r)
      coeffs[(nbase + mofs + r) * 4 + n] = C[r];
  }
}

// ---------------------------------------------------------------------------
// Kernel 3: y[i,j] = c0 + c1*a + c2*b + c3*a*b with a=x[i,ia], b=x[i,ib].
// Per-neuron params live in registers for a 64-row batch chunk; x (33.5 MB)
// stays L2-resident, gathers hit L2/WGP$; y stores are coalesced and
// nontemporal (streaming 134 MB, don't evict x from L2). global_prefetch_b8
// pulls next row's gather targets.
// ---------------------------------------------------------------------------
__global__ __launch_bounds__(256) void logic_main_kernel(
    const float* __restrict__ x,
    const int*   __restrict__ idx_a,
    const int*   __restrict__ idx_b,
    const f32x4* __restrict__ coeffs,
    float* __restrict__ y) {
  const int j  = blockIdx.x * blockDim.x + threadIdx.x;  // neuron (coalesced)
  const int i0 = blockIdx.y * ROWS_PER_BLOCK;            // batch chunk start
  const int ia = idx_a[j];
  const int ib = idx_b[j];
  const f32x4 c = coeffs[j];

  const float* xr = x + (size_t)i0 * IN_DIM;
  float*       yr = y + (size_t)i0 * OUT_DIM + j;

  #pragma unroll 4
  for (int r = 0; r < ROWS_PER_BLOCK; ++r) {
    __builtin_prefetch(xr + IN_DIM + ia, 0, 0);  // next row's gather targets
    __builtin_prefetch(xr + IN_DIM + ib, 0, 0);
    const float a = xr[ia];
    const float b = xr[ib];
    const float v = fmaf(c.w, a * b, fmaf(c.z, b, fmaf(c.y, a, c.x)));
    __builtin_nontemporal_store(v, yr);
    xr += IN_DIM;
    yr += OUT_DIM;
  }
}

// ---------------------------------------------------------------------------
// Launch. Workspace layout (needs 192 KB):
//   [0, 32KB)      idx_a  : OUT_DIM int
//   [32KB, 64KB)   idx_b  : OUT_DIM int
//   [64KB, 192KB)  coeffs : OUT_DIM float4
// ---------------------------------------------------------------------------
extern "C" void kernel_launch(void* const* d_in, const int* in_sizes, int n_in,
                              void* d_out, int out_size, void* d_ws, size_t ws_size,
                              hipStream_t stream) {
  const float*         x     = (const float*)d_in[0];
  const float*         nw    = (const float*)d_in[1];
  const float*         wa    = (const float*)d_in[2];
  const float*         wb    = (const float*)d_in[3];
  const unsigned char* gmask = (const unsigned char*)d_in[4];
  const unsigned char* ma    = (const unsigned char*)d_in[5];
  const unsigned char* mb    = (const unsigned char*)d_in[6];
  float* y = (float*)d_out;

  int*   idx_a  = (int*)d_ws;
  int*   idx_b  = idx_a + OUT_DIM;
  f32x4* coeffs = (f32x4*)(idx_b + OUT_DIM);

  // 2*OUT_DIM rows, one wave32 each, 8 waves per 256-thread block
  link_argmax_kernel<<<(2 * OUT_DIM) / 8, 256, 0, stream>>>(
      wa, wb, ma, mb, idx_a, idx_b);

  // OUT_DIM/16 waves of 32 lanes -> 16384 threads -> 64 blocks
  gate_wmma_kernel<<<(OUT_DIM / 16) * 32 / 256, 256, 0, stream>>>(
      nw, gmask, (float*)coeffs);

  dim3 grid(OUT_DIM / 256, BATCH / ROWS_PER_BLOCK);
  logic_main_kernel<<<grid, 256, 0, stream>>>(x, idx_a, idx_b, coeffs, y);
}
